// BNBLinear4bit_87574383165510
// MI455X (gfx1250) — matrix operand order repaired and checked by reading, not compiled, not run on hardware
//
#include <hip/hip_runtime.h>
#include <hip/hip_fp16.h>

typedef __attribute__((ext_vector_type(16))) _Float16 v16h;
typedef __attribute__((ext_vector_type(2)))  _Float16 v2h;
typedef __attribute__((ext_vector_type(8)))  float    v8f;

#define BM 256
#define BN 128
#define BK 32
#define LDT 40   // padded LDS row stride in 16-bit elements (80 B, 16B-aligned)

__device__ __constant__ float NF4_TBL[16] = {
    -1.0f, -0.6961928009986877f, -0.5250730514526367f, -0.39491748809814453f,
    -0.28444138169288635f, -0.18477343022823334f, -0.09105003625154495f, 0.0f,
    0.07958029955625534f, 0.16093020141124725f, 0.24611230194568634f,
    0.33791524171829224f, 0.44070982933044434f, 0.5626170039176941f,
    0.6797559261322021f, 1.0f};

union FragF16 { uint4 q[2]; v16h v; };

static __device__ __forceinline__ unsigned int cvt2(float a, float b) {
    return __builtin_bit_cast(unsigned int, __builtin_amdgcn_cvt_pkrtz(a, b));
}

// dequantize a pair of int32 NF4 codes to packed f16, scaled
static __device__ __forceinline__ unsigned int deq_pair(int c0, int c1, int cb16, v2h s2) {
    int l0 = __builtin_amdgcn_ds_bpermute(c0 << 2, cb16);   // f16 code value in low 16
    int l1 = __builtin_amdgcn_ds_bpermute(c1 << 2, cb16);
    unsigned int p = __builtin_amdgcn_perm((unsigned)l1, (unsigned)l0, 0x05040100u);
    v2h w = __builtin_bit_cast(v2h, p) * s2;                // v_pk_mul_f16
    return __builtin_bit_cast(unsigned int, w);
}

struct Staged {
    float4 fa[2][4];   // x: 2 rows x 16 floats
    int4   cw[4];      // codes: 16 codes
    float  scale;
};

__global__ __launch_bounds__(256)
void nf4_wmma_gemm(const float* __restrict__ x,
                   const int*   __restrict__ codes,
                   const float* __restrict__ absmax,
                   const float* __restrict__ bias,
                   float*       __restrict__ out,
                   int M, int N, int K)
{
    __shared__ __align__(16) unsigned short Ash[2][BM * LDT];  // x tiles, f16 [m][k]
    __shared__ __align__(16) unsigned short Wsh[2][BN * LDT];  // dequant W tiles, f16 [n][k]

    const int tid  = threadIdx.x;
    const int lane = tid & 31;
    const int wave = tid >> 5;
    const int wm   = wave & 3;      // 4 waves tile M (64 rows each)
    const int wn   = wave >> 2;     // 2 waves tile N (64 cols each)
    const int h    = lane >> 4;     // half-wave selector (K-half per WMMA layout)
    const int ln   = lane & 15;

    const int n0 = blockIdx.x * BN;
    const int m0 = blockIdx.y * BM;

    // per-lane NF4 codebook entry as f16 bits (ds_bpermute source)
    union { _Float16 hh[2]; int ii; } cbu;
    cbu.hh[0] = (_Float16)NF4_TBL[ln];
    cbu.hh[1] = (_Float16)0.0f;
    const int cb16 = cbu.ii;

    // A loader: thread t -> rows (t>>1) and (t>>1)+128, k-half (t&1)*16 (16 floats)
    const int arow  = tid >> 1;
    const int akoff = (tid & 1) * 16;
    // W loader: thread t -> row t>>1, k-half (t&1)*16 (16 codes)
    const int wrow  = arow;
    const int wkoff = akoff;
    const int kb    = K >> 6;       // absmax blocks per weight row (BLOCKSIZE=64)

    const float* xbase = x + (size_t)(m0 + arow) * K + akoff;
    const int*   cbase = codes + (size_t)(n0 + wrow) * K + wkoff;
    const size_t xrow2 = (size_t)128 * K;   // second A row offset

    v8f acc[4][4] = {};
    Staged s;

    // ---------- prologue: load + stage tile k0=0 into buffer 0 ----------
    {
        const float4* xp0 = (const float4*)(xbase);
        const float4* xp1 = (const float4*)(xbase + xrow2);
        const int4*   cp  = (const int4*)(cbase);
#pragma unroll
        for (int j = 0; j < 4; ++j) { s.fa[0][j] = xp0[j]; s.fa[1][j] = xp1[j]; s.cw[j] = cp[j]; }
        s.scale = absmax[(size_t)(n0 + wrow) * kb];
    }
    {
#pragma unroll
        for (int rr = 0; rr < 2; ++rr) {
            uint4 u0, u1;
            u0.x = cvt2(s.fa[rr][0].x, s.fa[rr][0].y); u0.y = cvt2(s.fa[rr][0].z, s.fa[rr][0].w);
            u0.z = cvt2(s.fa[rr][1].x, s.fa[rr][1].y); u0.w = cvt2(s.fa[rr][1].z, s.fa[rr][1].w);
            u1.x = cvt2(s.fa[rr][2].x, s.fa[rr][2].y); u1.y = cvt2(s.fa[rr][2].z, s.fa[rr][2].w);
            u1.z = cvt2(s.fa[rr][3].x, s.fa[rr][3].y); u1.w = cvt2(s.fa[rr][3].z, s.fa[rr][3].w);
            uint4* dst = (uint4*)&Ash[0][(arow + 128 * rr) * LDT + akoff];
            dst[0] = u0; dst[1] = u1;
        }
        const _Float16 sh = (_Float16)s.scale;
        const v2h s2 = {sh, sh};
        uint4 w0, w1;
        w0.x = deq_pair(s.cw[0].x, s.cw[0].y, cb16, s2);
        w0.y = deq_pair(s.cw[0].z, s.cw[0].w, cb16, s2);
        w0.z = deq_pair(s.cw[1].x, s.cw[1].y, cb16, s2);
        w0.w = deq_pair(s.cw[1].z, s.cw[1].w, cb16, s2);
        w1.x = deq_pair(s.cw[2].x, s.cw[2].y, cb16, s2);
        w1.y = deq_pair(s.cw[2].z, s.cw[2].w, cb16, s2);
        w1.z = deq_pair(s.cw[3].x, s.cw[3].y, cb16, s2);
        w1.w = deq_pair(s.cw[3].z, s.cw[3].w, cb16, s2);
        uint4* dst = (uint4*)&Wsh[0][wrow * LDT + wkoff];
        dst[0] = w0; dst[1] = w1;
    }
    __syncthreads();

    // ---------- main pipelined loop ----------
    int buf = 0;
    for (int k0 = 0; k0 < K; k0 += BK) {
        const bool more = (k0 + BK) < K;

        // phase 1: issue next tile's global loads into registers
        if (more) {
            const int kn = k0 + BK;
            const float4* xp0 = (const float4*)(xbase + kn);
            const float4* xp1 = (const float4*)(xbase + xrow2 + kn);
            const int4*   cp  = (const int4*)(cbase + kn);
            __builtin_prefetch((const char*)xp0 + BK * 4, 0, 0);     // k0 + 2*BK
            __builtin_prefetch((const char*)cp  + BK * 4, 0, 0);
#pragma unroll
            for (int j = 0; j < 4; ++j) { s.fa[0][j] = xp0[j]; s.fa[1][j] = xp1[j]; s.cw[j] = cp[j]; }
            s.scale = absmax[(size_t)(n0 + wrow) * kb + (kn >> 6)];
        }

        // phase 2: compute from current buffer (WMMA hides the loads above)
        {
            FragF16 bfrag[4];
#pragma unroll
            for (int ni = 0; ni < 4; ++ni) {
                // B 32x16 fragment, contiguous k order (same as A): lane ln = col N; h picks K range {0..15}/{16..31}
                const unsigned short* bp = &Wsh[buf][(wn * 64 + ni * 16 + ln) * LDT + 16 * h];
                bfrag[ni].q[0] = *(const uint4*)(bp);
                bfrag[ni].q[1] = *(const uint4*)(bp + 8);
            }
#pragma unroll
            for (int mi = 0; mi < 4; ++mi) {
                // A 16x32 fragment, contiguous k order (same as B): lane ln = row M; h picks K range {0..15}/{16..31}
                const unsigned short* ap = &Ash[buf][(wm * 64 + mi * 16 + ln) * LDT + 16 * h];
                FragF16 af;
                af.q[0] = *(const uint4*)(ap);
                af.q[1] = *(const uint4*)(ap + 8);
#pragma unroll
                for (int ni = 0; ni < 4; ++ni)
                    acc[mi][ni] = __builtin_amdgcn_wmma_f32_16x16x32_f16(
                        false, af.v, false, bfrag[ni].v,
                        (short)0, acc[mi][ni], false, false);
            }
        }

        // phase 3: convert + stage next tile into the other buffer
        if (more) {
            const int nb = buf ^ 1;
#pragma unroll
            for (int rr = 0; rr < 2; ++rr) {
                uint4 u0, u1;
                u0.x = cvt2(s.fa[rr][0].x, s.fa[rr][0].y); u0.y = cvt2(s.fa[rr][0].z, s.fa[rr][0].w);
                u0.z = cvt2(s.fa[rr][1].x, s.fa[rr][1].y); u0.w = cvt2(s.fa[rr][1].z, s.fa[rr][1].w);
                u1.x = cvt2(s.fa[rr][2].x, s.fa[rr][2].y); u1.y = cvt2(s.fa[rr][2].z, s.fa[rr][2].w);
                u1.z = cvt2(s.fa[rr][3].x, s.fa[rr][3].y); u1.w = cvt2(s.fa[rr][3].z, s.fa[rr][3].w);
                uint4* dst = (uint4*)&Ash[nb][(arow + 128 * rr) * LDT + akoff];
                dst[0] = u0; dst[1] = u1;
            }
            const _Float16 sh = (_Float16)s.scale;
            const v2h s2 = {sh, sh};
            uint4 w0, w1;
            w0.x = deq_pair(s.cw[0].x, s.cw[0].y, cb16, s2);
            w0.y = deq_pair(s.cw[0].z, s.cw[0].w, cb16, s2);
            w0.z = deq_pair(s.cw[1].x, s.cw[1].y, cb16, s2);
            w0.w = deq_pair(s.cw[1].z, s.cw[1].w, cb16, s2);
            w1.x = deq_pair(s.cw[2].x, s.cw[2].y, cb16, s2);
            w1.y = deq_pair(s.cw[2].z, s.cw[2].w, cb16, s2);
            w1.z = deq_pair(s.cw[3].x, s.cw[3].y, cb16, s2);
            w1.w = deq_pair(s.cw[3].z, s.cw[3].w, cb16, s2);
            uint4* dst = (uint4*)&Wsh[nb][wrow * LDT + wkoff];
            dst[0] = w0; dst[1] = w1;
        }

        __syncthreads();
        buf ^= 1;
    }

    // ---------- epilogue: bias add, nan/inf -> 0, store f32 ----------
#pragma unroll
    for (int mi = 0; mi < 4; ++mi) {
#pragma unroll
        for (int ni = 0; ni < 4; ++ni) {
            const int n     = n0 + wn * 64 + ni * 16 + ln;
            const int mbase = m0 + wm * 64 + mi * 16 + h * 8;
            const float bv  = bias[n];
#pragma unroll
            for (int r = 0; r < 8; ++r) {
                float v = acc[mi][ni][r] + bv;
                if ((__float_as_uint(v) & 0x7F800000u) == 0x7F800000u) v = 0.0f;
                out[(size_t)(mbase + r) * N + n] = v;
            }
        }
    }
}

extern "C" void kernel_launch(void* const* d_in, const int* in_sizes, int n_in,
                              void* d_out, int out_size, void* d_ws, size_t ws_size,
                              hipStream_t stream) {
    const float* x      = (const float*)d_in[0];
    const int*   codes  = (const int*)d_in[1];
    const float* absmax = (const float*)d_in[2];
    const float* bias   = (const float*)d_in[3];
    float*       out    = (float*)d_out;

    const int N = in_sizes[3];                 // OUT = 4096
    const int K = in_sizes[1] / N;             // IN  = 4096
    const int M = in_sizes[0] / K;             // B*S = 4096

    dim3 grid((unsigned)(N / BN), (unsigned)(M / BM));
    dim3 block(256);
    nf4_wmma_gemm<<<grid, block, 0, stream>>>(x, codes, absmax, bias, out, M, N, K);
}
